// FFGAT_47605417509433
// MI455X (gfx1250) — matrix operand order repaired and checked by reading, not compiled, not run on hardware
//
#include <hip/hip_runtime.h>

#define BB 16
#define NN 1024
#define FF 128
#define HH 8
#define DD 16
#define SLOPE 0.2f
#define LN_EPS 1e-5f

typedef __attribute__((ext_vector_type(16))) _Float16 v16h;
typedef __attribute__((ext_vector_type(8)))  _Float16 v8h;
typedef __attribute__((ext_vector_type(4)))  _Float16 v4h;
typedef __attribute__((ext_vector_type(2)))  __fp16   v2fp;   // cvt_pkrtz result type
typedef __attribute__((ext_vector_type(8)))  float    v8f;

__device__ __forceinline__ float lrelu(float v) { return fmaxf(v, SLOPE * v); }

// ---------------------------------------------------------------------------
// Pack adj int32 (0/1) -> bitmask, one 32-bit word per 32 consecutive j.
// 32x HBM traffic reduction for the attention passes (64MB -> 2MB resident).
// ---------------------------------------------------------------------------
__global__ __launch_bounds__(256) void ffgat_pack_adj(
    const int* __restrict__ adj, unsigned* __restrict__ amask)
{
    const size_t t = (size_t)blockIdx.x * 256 + threadIdx.x;  // over B*N*N
    const int pred = adj[t] > 0;
    const unsigned w = (unsigned)__ballot(pred);
    if ((threadIdx.x & 31) == 0) amask[t >> 5] = w;
}

// ---------------------------------------------------------------------------
// LayerNorm over F=128 features, one wave (32 lanes x 4 floats) per row.
// ---------------------------------------------------------------------------
__global__ __launch_bounds__(256) void ffgat_ln_f16(
    const float* __restrict__ x, const float* __restrict__ g,
    const float* __restrict__ bta, _Float16* __restrict__ out)
{
    const int row  = blockIdx.x * 8 + (threadIdx.x >> 5);
    const int lane = threadIdx.x & 31;
    const float4 v = *(const float4*)(x + (size_t)row * FF + lane * 4);
    float s = v.x + v.y + v.z + v.w;
#pragma unroll
    for (int o = 16; o >= 1; o >>= 1) s += __shfl_xor(s, o);
    const float mu = s * (1.0f / FF);
    const float dx = v.x - mu, dy = v.y - mu, dz = v.z - mu, dw = v.w - mu;
    float ss = dx * dx + dy * dy + dz * dz + dw * dw;
#pragma unroll
    for (int o = 16; o >= 1; o >>= 1) ss += __shfl_xor(ss, o);
    const float inv = rsqrtf(ss * (1.0f / FF) + LN_EPS);
    const float4 gv = *(const float4*)(g + lane * 4);
    const float4 bv = *(const float4*)(bta + lane * 4);
    v4h o;
    o[0] = (_Float16)(gv.x * dx * inv + bv.x);
    o[1] = (_Float16)(gv.y * dy * inv + bv.y);
    o[2] = (_Float16)(gv.z * dz * inv + bv.z);
    o[3] = (_Float16)(gv.w * dw * inv + bv.w);
    *(v4h*)(out + (size_t)row * FF + lane * 4) = o;
}

// ---------------------------------------------------------------------------
// W [K=128][N=128] f32  ->  WT [N][K] f16 (B-matrix friendly: K contiguous)
// ---------------------------------------------------------------------------
__global__ void ffgat_prep_w(const float* __restrict__ W, _Float16* __restrict__ WT)
{
    const int n = blockIdx.x, k = threadIdx.x;
    WT[n * FF + k] = (_Float16)W[k * FF + n];
}

// ---------------------------------------------------------------------------
// C[M,128] = A[M,128](f16) @ W(from WT f16) + bias, optional ReLU.
// 8 waves per block; wave w owns output-column tile w. 4x WMMA K-loop.
// ---------------------------------------------------------------------------
__global__ __launch_bounds__(256) void ffgat_gemm16(
    const _Float16* __restrict__ A, const _Float16* __restrict__ WT,
    const float* __restrict__ bias, float* __restrict__ C, int relu)
{
    const int i0   = blockIdx.x * 16;
    const int n0   = (threadIdx.x >> 5) * 16;
    const int lane = threadIdx.x & 31;
    const int low  = lane & 15, hi = lane >> 4;
    const _Float16* Ar = A + (size_t)(i0 + low) * FF + hi * 16;
    const _Float16* Wr = WT + (size_t)(n0 + low) * FF + hi * 16;

    v8f acc = {0.f, 0.f, 0.f, 0.f, 0.f, 0.f, 0.f, 0.f};
#pragma unroll
    for (int kt = 0; kt < FF / 32; ++kt) {
        const int k0 = kt * 32;
        const v8h a0 = *(const v8h*)(Ar + k0);
        const v8h a1 = *(const v8h*)(Ar + k0 + 8);
        v16h a;
#pragma unroll
        for (int q = 0; q < 8; ++q) { a[q] = a0[q]; a[q + 8] = a1[q]; }
        const v16h bm = *(const v16h*)(Wr + k0);
        acc = __builtin_amdgcn_wmma_f32_16x16x32_f16(false, a, false, bm,
                                                     (short)0, acc, false, false);
    }
    const int col = n0 + low;
    const float bv = bias ? bias[col] : 0.0f;
#pragma unroll
    for (int r = 0; r < 8; ++r) {
        float v = acc[r] + bv;
        if (relu) v = fmaxf(v, 0.0f);
        C[(size_t)(i0 + r + hi * 8) * FF + col] = v;
    }
}

// ---------------------------------------------------------------------------
// From h[B,N,H*D] f32: per (b,n,h) compute s,t scores, exp(t), exp(0.2t),
// and transposed f16 hT[b][h][d][n] (WMMA B-matrix layout for aggregation).
// ---------------------------------------------------------------------------
__global__ __launch_bounds__(256) void ffgat_scores(
    const float* __restrict__ hmat, const float* __restrict__ a_src,
    const float* __restrict__ a_dst, float* __restrict__ sT,
    float* __restrict__ tT, float* __restrict__ tE1, float* __restrict__ tE2,
    _Float16* __restrict__ hT)
{
    const int t = blockIdx.x * 256 + threadIdx.x;   // 0 .. B*H*N-1
    const int n = t & (NN - 1);
    const int h = (t >> 10) & (HH - 1);
    const int b = t >> 13;
    const float* hp = hmat + (size_t)(b * NN + n) * FF + h * DD;
    float hv[DD];
#pragma unroll
    for (int d = 0; d < DD; ++d) hv[d] = hp[d];
    float s = 0.f, tv = 0.f;
#pragma unroll
    for (int d = 0; d < DD; ++d) {
        s  += hv[d] * a_src[h * DD + d];
        tv += hv[d] * a_dst[h * DD + d];
    }
    const size_t bh = (size_t)(b * HH + h);
    sT [bh * NN + n] = s;
    tT [bh * NN + n] = tv;
    tE1[bh * NN + n] = __expf(tv);
    tE2[bh * NN + n] = __expf(SLOPE * tv);
    _Float16* hTp = hT + bh * DD * NN + n;
#pragma unroll
    for (int d = 0; d < DD; ++d) hTp[(size_t)d * NN] = (_Float16)hv[d];
}

// ---------------------------------------------------------------------------
// T[b,h] = max_j t[b,h,j]  (exact unmasked row-max seed via lrelu monotonicity)
// ---------------------------------------------------------------------------
__global__ __launch_bounds__(256) void ffgat_tmax(const float* __restrict__ tT,
                                                  float* __restrict__ Tmax)
{
    const int bh = blockIdx.x;
    const int tid = threadIdx.x;
    float m = -1e30f;
    for (int j = tid; j < NN; j += 256) m = fmaxf(m, tT[(size_t)bh * NN + j]);
#pragma unroll
    for (int o = 16; o >= 1; o >>= 1) m = fmaxf(m, __shfl_xor(m, o));
    __shared__ float red[8];
    if ((tid & 31) == 0) red[tid >> 5] = m;
    __syncthreads();
    if (tid == 0) {
        float mm = red[0];
        for (int w = 1; w < 8; ++w) mm = fmaxf(mm, red[w]);
        Tmax[bh] = mm;
    }
}

// ---------------------------------------------------------------------------
// Flash-style masked-softmax aggregation. Block = (b, 16-row i-tile),
// 8 independent waves = 8 heads, no LDS/barriers. Stream j in 32-tiles:
//   p_ij = adjbit ? ((exp(t) > exp(-s)) ? A1*exp(t) : A2*exp(0.2t)) : 0
// adjacency from packed bitmask (one broadcast b32 per lane per tile);
// p pairs packed with v_cvt_pkrtz into the WMMA A registers.
// ---------------------------------------------------------------------------
__global__ __launch_bounds__(256) void ffgat_attn(
    const float* __restrict__ sT, const float* __restrict__ tE1,
    const float* __restrict__ tE2, const float* __restrict__ Tmax,
    const unsigned* __restrict__ amask, const _Float16* __restrict__ hT,
    float* __restrict__ out)
{
    const int b    = blockIdx.x;
    const int i0   = blockIdx.y * 16;
    const int h    = threadIdx.x >> 5;
    const int lane = threadIdx.x & 31;
    const int low  = lane & 15, hi = lane >> 4;
    const int iL   = i0 + low;                     // this lane's A-matrix row
    const size_t bh = (size_t)(b * HH + h);

    const float s  = sT[bh * NN + iL];
    const float M  = lrelu(s + Tmax[bh]);
    const float A1 = __expf(s - M);
    const float A2 = __expf(SLOPE * s - M);
    const float Ct = __expf(-s);                   // e1 > Ct  <=>  s + t > 0

    const float* e1p = tE1 + bh * NN;
    const float* e2p = tE2 + bh * NN;
    const _Float16* hp = hT + bh * DD * NN + (size_t)low * NN + hi * 16;
    const unsigned* ap = amask + ((size_t)b * NN + iL) * (NN / 32);
    const int kb = hi * 16;                        // A K-offset base per lane half

    v8f acc = {0.f, 0.f, 0.f, 0.f, 0.f, 0.f, 0.f, 0.f};
    float den = 0.f;

    union AV { v16h v; v2fp p[8]; } av;

    for (int jt = 0; jt < NN / 32; ++jt) {
        const int j0 = jt * 32;
        const unsigned wl = ap[jt] >> kb;          // align lane's bits to const pos
#pragma unroll
        for (int g = 0; g < 2; ++g) {
            const int jb = j0 + g * 8 + kb;
            const float4 e1a = *(const float4*)(e1p + jb);
            const float4 e1b = *(const float4*)(e1p + jb + 4);
            const float4 e2a = *(const float4*)(e2p + jb);
            const float4 e2b = *(const float4*)(e2p + jb + 4);
            const float ev1[8] = {e1a.x, e1a.y, e1a.z, e1a.w, e1b.x, e1b.y, e1b.z, e1b.w};
            const float ev2[8] = {e2a.x, e2a.y, e2a.z, e2a.w, e2b.x, e2b.y, e2b.z, e2b.w};
#pragma unroll
            for (int q2 = 0; q2 < 4; ++q2) {
                const int q0 = 2 * q2, q1 = 2 * q2 + 1;
                const float pe0 = (ev1[q0] > Ct) ? (A1 * ev1[q0]) : (A2 * ev2[q0]);
                const float pe1 = (ev1[q1] > Ct) ? (A1 * ev1[q1]) : (A2 * ev2[q1]);
                const float p0 = (wl & (1u << (g * 8 + q0))) ? pe0 : 0.0f;
                const float p1 = (wl & (1u << (g * 8 + q1))) ? pe1 : 0.0f;
                den += p0 + p1;
                av.p[g * 4 + q2] = __builtin_amdgcn_cvt_pkrtz(p0, p1);
            }
        }
        const v16h bm = *(const v16h*)(hp + j0);   // 16 contiguous halves: B-layout
        acc = __builtin_amdgcn_wmma_f32_16x16x32_f16(false, av.v, false, bm,
                                                     (short)0, acc, false, false);
    }

    den += __shfl_xor(den, 16);                    // full row-sum on every lane
    den = fmaxf(den, 1e-30f);

    float* op = out + (size_t)b * NN * FF + h * DD + low;
#pragma unroll
    for (int r = 0; r < 8; ++r) {
        const float dlo = __shfl(den, r);          // den of row i0+r
        const float dhi = __shfl(den, r + 8);      // den of row i0+r+8
        const float dd  = hi ? dhi : dlo;
        op[(size_t)(i0 + r + hi * 8) * FF] = acc[r] / dd;
    }
}

// ---------------------------------------------------------------------------
extern "C" void kernel_launch(void* const* d_in, const int* in_sizes, int n_in,
                              void* d_out, int out_size, void* d_ws, size_t ws_size,
                              hipStream_t stream)
{
    (void)in_sizes; (void)n_in; (void)out_size; (void)ws_size;

    const float* x      = (const float*)d_in[0];
    const int*   adj    = (const int*)  d_in[1];
    const float* ln1_g  = (const float*)d_in[2];
    const float* ln1_b  = (const float*)d_in[3];
    const float* W1     = (const float*)d_in[4];
    const float* a_src1 = (const float*)d_in[5];
    const float* a_dst1 = (const float*)d_in[6];
    const float* ln2_g  = (const float*)d_in[7];
    const float* ln2_b  = (const float*)d_in[8];
    const float* fc_w   = (const float*)d_in[9];
    const float* fc_b   = (const float*)d_in[10];
    const float* ln3_g  = (const float*)d_in[11];
    const float* ln3_b  = (const float*)d_in[12];
    const float* W2     = (const float*)d_in[13];
    const float* a_src2 = (const float*)d_in[14];
    const float* a_dst2 = (const float*)d_in[15];
    float* outp = (float*)d_out;

    char* ws = (char*)d_ws;
    size_t off = 0;
    auto carve = [&](size_t bytes) -> char* {
        char* p = ws + off;
        off += (bytes + 255) & ~(size_t)255;
        return p;
    };
    _Float16* lnbuf  = (_Float16*)carve((size_t)BB * NN * FF * 2);       // 4 MB
    float*    hmat   = (float*)   carve((size_t)BB * NN * FF * 4);       // 8 MB
    float*    gatout = (float*)   carve((size_t)BB * NN * FF * 4);       // 8 MB
    _Float16* hT     = (_Float16*)carve((size_t)BB * HH * DD * NN * 2);  // 4 MB
    unsigned* amask  = (unsigned*)carve((size_t)BB * NN * (NN / 32) * 4);// 2 MB
    float*    sT     = (float*)   carve((size_t)BB * HH * NN * 4);
    float*    tT     = (float*)   carve((size_t)BB * HH * NN * 4);
    float*    tE1    = (float*)   carve((size_t)BB * HH * NN * 4);
    float*    tE2    = (float*)   carve((size_t)BB * HH * NN * 4);
    float*    Tm     = (float*)   carve((size_t)BB * HH * 4);
    _Float16* WT1    = (_Float16*)carve((size_t)FF * FF * 2);
    _Float16* WTf    = (_Float16*)carve((size_t)FF * FF * 2);
    _Float16* WT2    = (_Float16*)carve((size_t)FF * FF * 2);

    // one-time prep: weight transposes + adjacency bit-pack
    ffgat_prep_w  <<<FF, FF, 0, stream>>>(W1,   WT1);
    ffgat_prep_w  <<<FF, FF, 0, stream>>>(fc_w, WTf);
    ffgat_prep_w  <<<FF, FF, 0, stream>>>(W2,   WT2);
    ffgat_pack_adj<<<(unsigned)((size_t)BB * NN * NN / 256), 256, 0, stream>>>(adj, amask);

    // ---- layer 1: LN1 -> h1 = ln(x)@W1 -> scores -> attention ----
    ffgat_ln_f16 <<<BB * NN / 8,  256, 0, stream>>>(x, ln1_g, ln1_b, lnbuf);
    ffgat_gemm16 <<<BB * NN / 16, 256, 0, stream>>>(lnbuf, WT1, nullptr, hmat, 0);
    ffgat_scores <<<BB * HH * NN / 256, 256, 0, stream>>>(hmat, a_src1, a_dst1,
                                                          sT, tT, tE1, tE2, hT);
    ffgat_tmax   <<<BB * HH, 256, 0, stream>>>(tT, Tm);
    ffgat_attn   <<<dim3(BB, NN / 16), 256, 0, stream>>>(sT, tE1, tE2, Tm, amask, hT, gatout);

    // ---- MLP: LN2 -> relu(fc) -> LN3 ----
    ffgat_ln_f16 <<<BB * NN / 8,  256, 0, stream>>>(gatout, ln2_g, ln2_b, lnbuf);
    ffgat_gemm16 <<<BB * NN / 16, 256, 0, stream>>>(lnbuf, WTf, fc_b, hmat, 1);
    ffgat_ln_f16 <<<BB * NN / 8,  256, 0, stream>>>(hmat, ln3_g, ln3_b, lnbuf);

    // ---- layer 2: h2 = ln3@W2 -> scores -> attention -> d_out ----
    ffgat_gemm16 <<<BB * NN / 16, 256, 0, stream>>>(lnbuf, WT2, nullptr, hmat, 0);
    ffgat_scores <<<BB * HH * NN / 256, 256, 0, stream>>>(hmat, a_src2, a_dst2,
                                                          sT, tT, tE1, tE2, hT);
    ffgat_tmax   <<<BB * HH, 256, 0, stream>>>(tT, Tm);
    ffgat_attn   <<<dim3(BB, NN / 16), 256, 0, stream>>>(sT, tE1, tE2, Tm, amask, hT, outp);
}
